// ToricCRBM_88244398063952
// MI455X (gfx1250) — compile-verified, matrix operand order checked by reading, not run on hardware
//
#include <hip/hip_runtime.h>
#include <hip/hip_bf16.h>
#include <math.h>

// ---------------------------------------------------------------------------
// ToricCRBM forward, fused as one big bf16 WMMA GEMM:
//   theta[b, f*S+s] = sum_k Xaug[b,k] * W[f*S+s, k] + hidden_bias[f]
//   Xaug = [ x (1152) | cv0 (576) | cv1 (576) | cv2 (24) | cv3 (24) | pad(16) ]
//   W    = gathered symm/corr kernels (batch-independent, built once)
// plus two loop-correlator WMMAs (K=24 padded to 32) per tile, fused
// log_cosh + reduction epilogue. Out initialized with the visible bias.
// ---------------------------------------------------------------------------

typedef __bf16 bf16_t;
typedef __attribute__((ext_vector_type(16))) __bf16 v16bf;
typedef __attribute__((ext_vector_type(8)))  float  v8f;
typedef __attribute__((ext_vector_type(4)))  int    v4i;

#define B_     16384
#define N_     1152
#define S_     576
#define F_     8
#define FS_    4608            // F_*S_
#define KTOT   2368            // 74 * 32 (2352 real + 16 pad)
#define KSTEPS 74
#define CT_    288             // FS_/16 column tiles
#define XU4    296             // uint4 per Xaug row (2368*2/16)
#define LOG2F_ 0.6931471805599453f

#if defined(__has_builtin)
#if __has_builtin(__builtin_amdgcn_global_load_async_to_lds_b128) && \
    __has_builtin(__builtin_amdgcn_s_wait_asynccnt)
#define USE_ASYNC_LDS 1
#endif
#endif

typedef __attribute__((address_space(1))) v4i gv4i;   // global int4
typedef __attribute__((address_space(3))) v4i lv4i;   // LDS int4

union Frag {
    uint4 u[2];
    v16bf v;
};

__device__ __forceinline__ float log_cosh_f(float t) {
    float a = fabsf(t);
    return a + log1pf(__expf(-2.0f * a)) - LOG2F_;
}

// ---------------------------------------------------------------------------
// Pack main weight matrix W[fs][k] (gathered) into per-lane WMMA B-fragment
// order: Wp[((ct*74+ks)*32+lane)*16 + j], j-th bf16 of lane's v16bf.
// k(j,h) = ks*32 + (j<8 ? 0 : 16) + h*8 + (j&7), h = lane>>4, n = lane&15.
// ---------------------------------------------------------------------------
__global__ __launch_bounds__(256)
void pack_w(const int* __restrict__ symm,
            const int* __restrict__ c0s, const int* __restrict__ c1s,
            const int* __restrict__ c2s, const int* __restrict__ c3s,
            const float* __restrict__ sk,
            const float* __restrict__ k0, const float* __restrict__ k1,
            const float* __restrict__ k2, const float* __restrict__ k3,
            bf16_t* __restrict__ Wp)
{
    int tid = blockIdx.x * 256 + threadIdx.x;
    if (tid >= CT_ * KSTEPS * 32) return;
    int lane = tid & 31;
    int ks   = (tid >> 5) % KSTEPS;
    int ct   = tid / (32 * KSTEPS);
    int fs   = ct * 16 + (lane & 15);
    int h    = lane >> 4;
    int f    = fs / S_;
    int s    = fs % S_;

    Frag out;
    bf16_t* ov = (bf16_t*)&out;
    #pragma unroll
    for (int j = 0; j < 16; ++j) {
        int k = ks * 32 + ((j & 8) << 1) + h * 8 + (j & 7);
        float v;
        if (k < 1152)      v = sk[f * 1152 + symm[s * 1152 + k]];
        else if (k < 1728) v = k0[f * 576 + c0s[s * 576 + (k - 1152)]];
        else if (k < 2304) v = k1[f * 576 + c1s[s * 576 + (k - 1728)]];
        else if (k < 2328) v = k2[f * 24  + c2s[s * 24  + (k - 2304)]];
        else if (k < 2352) v = k3[f * 24  + c3s[s * 24  + (k - 2328)]];
        else               v = 0.0f;
        ov[j] = (bf16_t)v;
    }
    uint4* dst = (uint4*)(Wp + (size_t)tid * 16);
    dst[0] = out.u[0];
    dst[1] = out.u[1];
}

// ---------------------------------------------------------------------------
// Pack the two loop-correlator weight matrices (K=24, padded to 32 with 0).
// Lp[(loop*CT_*32 + ct*32 + lane)*16 + j]
// ---------------------------------------------------------------------------
__global__ __launch_bounds__(256)
void pack_loops(const int* __restrict__ c2s, const int* __restrict__ c3s,
                const float* __restrict__ lk0, const float* __restrict__ lk1,
                bf16_t* __restrict__ Lp)
{
    int tid = blockIdx.x * 256 + threadIdx.x;
    if (tid >= 2 * CT_ * 32) return;
    int loop = tid / (CT_ * 32);
    int e    = tid % (CT_ * 32);
    int ct   = e / 32;
    int lane = e % 32;
    int fs   = ct * 16 + (lane & 15);
    int h    = lane >> 4;
    int f    = fs / S_;
    int s    = fs % S_;
    const int*   cs = loop ? c3s : c2s;
    const float* lk = loop ? lk1 : lk0;

    Frag out;
    bf16_t* ov = (bf16_t*)&out;
    #pragma unroll
    for (int j = 0; j < 16; ++j) {
        int k = ((j & 8) << 1) + h * 8 + (j & 7);
        float v = (k < 24) ? lk[f * 24 + cs[s * 24 + k]] : 0.0f;
        ov[j] = (bf16_t)v;
    }
    uint4* dst = (uint4*)(Lp + (size_t)tid * 16);
    dst[0] = out.u[0];
    dst[1] = out.u[1];
}

// ---------------------------------------------------------------------------
// Per batch row: build bf16 Xaug row (x | correlator products | pad),
// and initialize out[b] with the visible + correlator bias terms.
// ---------------------------------------------------------------------------
__global__ __launch_bounds__(256)
void prep_x(const float* __restrict__ x,
            const int* __restrict__ c0, const int* __restrict__ c1,
            const int* __restrict__ c2, const int* __restrict__ c3,
            const float* __restrict__ vb,
            const float* __restrict__ cb0, const float* __restrict__ cb1,
            const float* __restrict__ cb2, const float* __restrict__ cb3,
            bf16_t* __restrict__ Xa, float* __restrict__ out)
{
    __shared__ float xs[N_];
    __shared__ float sred;
    const int b   = blockIdx.x;
    const int tid = threadIdx.x;
    const float* xr = x + (size_t)b * N_;
    for (int k = tid; k < N_; k += 256) xs[k] = xr[k];
    if (tid == 0) sred = 0.0f;
    __syncthreads();

    bf16_t* Xrow = Xa + (size_t)b * KTOT;
    float acc = 0.0f;
    const float v0 = vb[0], v1 = vb[1];
    for (int i = tid; i < 576; i += 256) acc += xs[2 * i] * v0 + xs[2 * i + 1] * v1;
    for (int k = tid; k < N_; k += 256) Xrow[k] = (bf16_t)xs[k];

    const float bb0 = cb0[0], bb1 = cb1[0];
    for (int c = tid; c < 576; c += 256) {
        float p0 = xs[c0[4 * c]] * xs[c0[4 * c + 1]] * xs[c0[4 * c + 2]] * xs[c0[4 * c + 3]];
        float p1 = xs[c1[4 * c]] * xs[c1[4 * c + 1]] * xs[c1[4 * c + 2]] * xs[c1[4 * c + 3]];
        Xrow[1152 + c] = (bf16_t)p0;
        Xrow[1728 + c] = (bf16_t)p1;
        acc += bb0 * p0 + bb1 * p1;
    }
    if (tid < 24) {
        float p2 = 1.0f, p3 = 1.0f;
        #pragma unroll
        for (int j = 0; j < 24; ++j) {
            p2 *= xs[c2[tid * 24 + j]];
            p3 *= xs[c3[tid * 24 + j]];
        }
        Xrow[2304 + tid] = (bf16_t)p2;
        Xrow[2328 + tid] = (bf16_t)p3;
        acc += cb2[0] * p2 + cb3[0] * p3;
    }
    if (tid < 16) Xrow[2352 + tid] = (bf16_t)0.0f;

    atomicAdd(&sred, acc);
    __syncthreads();
    if (tid == 0) out[b] = sred;
}

// ---------------------------------------------------------------------------
// Main fused GEMM + log_cosh + reduction.
// Block = 256 threads = 8 wave32s. Block covers 16 batch rows x 128 FS cols.
// Each wave owns one 16x16 tile: 74 main WMMAs (two independent accumulator
// chains) + 2 loop WMMAs. A-tile staged via async global->LDS copy.
// ---------------------------------------------------------------------------
__global__ __launch_bounds__(256)
void crbm_gemm(const uint4* __restrict__ Xu4,
               const uint4* __restrict__ Wu4,
               const uint4* __restrict__ Lu4,
               const float* __restrict__ hidden_bias,
               const float* __restrict__ l0_bias,
               const float* __restrict__ l1_bias,
               float* __restrict__ out)
{
    __shared__ uint4 As[16 * XU4];   // 75776 B, one 16 x 2368 bf16 A row-tile
    __shared__ float rowAcc[16];

    const int tid     = threadIdx.x;
    const int rowBase = blockIdx.y * 16;

    const uint4* src = Xu4 + (size_t)rowBase * XU4;
#ifdef USE_ASYNC_LDS
    {
        // direct global -> LDS DMA, no VGPR round-trip; tracked by ASYNCcnt
        gv4i* gsrc = (gv4i*)src;
        lv4i* lds  = (lv4i*)As;
        for (int i = tid; i < 16 * XU4; i += 256) {
            __builtin_amdgcn_global_load_async_to_lds_b128(gsrc + i, lds + i, 0, 0);
        }
        __builtin_amdgcn_s_wait_asynccnt(0);
    }
#else
    for (int i = tid; i < 16 * XU4; i += 256) As[i] = src[i];
#endif
    if (tid < 16) rowAcc[tid] = 0.0f;
    __syncthreads();

    const int lane = tid & 31;
    const int w    = tid >> 5;
    const int ct   = blockIdx.x * 8 + w;      // column tile, [0, 288)
    const int r    = lane & 15;
    const int h    = lane >> 4;
    const int abase = r * XU4 + h;

    const uint4* wp = Wu4 + ((size_t)ct * KSTEPS * 32 + lane) * 2;

    v8f accA = {0.f, 0.f, 0.f, 0.f, 0.f, 0.f, 0.f, 0.f};
    v8f accB = {0.f, 0.f, 0.f, 0.f, 0.f, 0.f, 0.f, 0.f};
    v8f acc0 = {0.f, 0.f, 0.f, 0.f, 0.f, 0.f, 0.f, 0.f};
    v8f acc1 = {0.f, 0.f, 0.f, 0.f, 0.f, 0.f, 0.f, 0.f};
    Frag a0, b0, a1, b1;

    // two independent accumulator chains -> two WMMAs in flight per wave
    for (int ks = 0; ks < KSTEPS; ks += 2) {
        a0.u[0] = As[abase + ks * 4];
        a0.u[1] = As[abase + ks * 4 + 2];
        b0.u[0] = wp[0];
        b0.u[1] = wp[1];
        a1.u[0] = As[abase + ks * 4 + 4];
        a1.u[1] = As[abase + ks * 4 + 6];
        b1.u[0] = wp[64];
        b1.u[1] = wp[65];
        wp += 128;
        accA = __builtin_amdgcn_wmma_f32_16x16x32_bf16(
            false, a0.v, false, b0.v, (short)0, accA, false, false);
        accB = __builtin_amdgcn_wmma_f32_16x16x32_bf16(
            false, a1.v, false, b1.v, (short)0, accB, false, false);
    }

    // loop0: A window at half-offset 2304 (cv2 + cv3[0..7]; weight rows>=24 are 0)
    {
        const uint4* lp = Lu4 + ((size_t)(ct * 32 + lane)) * 2;
        a0.u[0] = As[abase + 288];
        a0.u[1] = As[abase + 290];
        b0.u[0] = lp[0];
        b0.u[1] = lp[1];
        acc0 = __builtin_amdgcn_wmma_f32_16x16x32_bf16(
            false, a0.v, false, b0.v, (short)0, acc0, false, false);
    }
    // loop1: A window at half-offset 2328 (cv3 + zero pad)
    {
        const uint4* lp = Lu4 + ((size_t)(CT_ * 32 + ct * 32 + lane)) * 2;
        a1.u[0] = As[abase + 291];
        a1.u[1] = As[abase + 293];
        b1.u[0] = lp[0];
        b1.u[1] = lp[1];
        acc1 = __builtin_amdgcn_wmma_f32_16x16x32_bf16(
            false, a1.v, false, b1.v, (short)0, acc1, false, false);
    }

    // Epilogue: biases, log_cosh, reduce over the 16 columns of the tile.
    const int   f  = ct / 36;                // 576/16 = 36 tiles per feature
    const float hb = hidden_bias[f];
    const float bl0 = l0_bias[f];
    const float bl1 = l1_bias[f];

    float lc[8];
    #pragma unroll
    for (int i = 0; i < 8; ++i) {
        lc[i] = log_cosh_f(accA[i] + accB[i] + hb)
              + log_cosh_f(acc0[i] + bl0)
              + log_cosh_f(acc1[i] + bl1);
    }
    // sum over the 16 lanes holding the same row group (xor within 16-lane half)
    #pragma unroll
    for (int m = 1; m < 16; m <<= 1) {
        #pragma unroll
        for (int i = 0; i < 8; ++i) lc[i] += __shfl_xor(lc[i], m, 32);
    }
    if ((lane & 15) == 0) {
        #pragma unroll
        for (int i = 0; i < 8; ++i) atomicAdd(&rowAcc[h * 8 + i], lc[i]);
    }
    __syncthreads();
    if (tid < 16) atomicAdd(out + rowBase + tid, rowAcc[tid]);
}

// ---------------------------------------------------------------------------
extern "C" void kernel_launch(void* const* d_in, const int* in_sizes, int n_in,
                              void* d_out, int out_size, void* d_ws, size_t ws_size,
                              hipStream_t stream)
{
    const float* x     = (const float*)d_in[0];
    const int*   symm  = (const int*)d_in[1];
    const int*   c0    = (const int*)d_in[2];
    const int*   c1    = (const int*)d_in[3];
    const int*   c2    = (const int*)d_in[4];
    const int*   c3    = (const int*)d_in[5];
    const int*   c0s   = (const int*)d_in[6];
    const int*   c1s   = (const int*)d_in[7];
    const int*   c2s   = (const int*)d_in[8];
    const int*   c3s   = (const int*)d_in[9];
    const float* hb    = (const float*)d_in[10];
    const float* sk    = (const float*)d_in[11];
    const float* vb    = (const float*)d_in[12];
    const float* cb0   = (const float*)d_in[13];
    const float* k0    = (const float*)d_in[14];
    const float* cb1   = (const float*)d_in[15];
    const float* k1    = (const float*)d_in[16];
    const float* cb2   = (const float*)d_in[17];
    const float* k2    = (const float*)d_in[18];
    const float* cb3   = (const float*)d_in[19];
    const float* k3    = (const float*)d_in[20];
    const float* l0hb  = (const float*)d_in[21];
    const float* lk0   = (const float*)d_in[22];
    const float* l1hb  = (const float*)d_in[23];
    const float* lk1   = (const float*)d_in[24];
    float* out = (float*)d_out;

    char* ws = (char*)d_ws;
    const size_t szW = (size_t)FS_ * KTOT * sizeof(bf16_t);        // 21,823,488
    const size_t szL = (size_t)2 * CT_ * 32 * 16 * sizeof(bf16_t); //    589,824
    bf16_t* Wp = (bf16_t*)(ws);
    bf16_t* Lp = (bf16_t*)(ws + szW);
    bf16_t* Xa = (bf16_t*)(ws + szW + szL);                        // 77,594,624

    // 1) build gathered weight matrices (batch independent)
    {
        int n = CT_ * KSTEPS * 32;
        pack_w<<<(n + 255) / 256, 256, 0, stream>>>(symm, c0s, c1s, c2s, c3s,
                                                    sk, k0, k1, k2, k3, Wp);
    }
    {
        int n = 2 * CT_ * 32;
        pack_loops<<<(n + 255) / 256, 256, 0, stream>>>(c2s, c3s, lk0, lk1, Lp);
    }
    // 2) augmented input rows + bias init of out
    prep_x<<<B_, 256, 0, stream>>>(x, c0, c1, c2, c3, vb, cb0, cb1, cb2, cb3,
                                   Xa, out);
    // 3) fused GEMM + log_cosh + reduction
    dim3 grid(CT_ / 8, B_ / 16);   // (36, 1024)
    crbm_gemm<<<grid, 256, 0, stream>>>((const uint4*)Xa, (const uint4*)Wp,
                                        (const uint4*)Lp, hb, l0hb, l1hb, out);
}
